// AlignmentLossWithSinkhorn_21973052686897
// MI455X (gfx1250) — matrix operand order repaired and checked by reading, not compiled, not run on hardware
//
#include <hip/hip_runtime.h>

#define NN 8192          // N == M (batch)
#define DD 64            // embedding dim
#define EPS_INV 20.0f    // 1/0.05
#define ITERS 20
#define SLICES 32        // row slices for column-sum partials
#define ROWS_PER_SLICE (NN / SLICES)
#define CHUNKS (NN / 32) // 256 k-chunks of 32 for final GEMM
#define SEGS 8           // m-range segments for final GEMM partials

typedef __attribute__((ext_vector_type(16))) _Float16 v16h;
typedef __attribute__((ext_vector_type(8)))  float    v8f;

// ---------------------------------------------------------------------------
// K0: v = 1
__global__ void sink_init_v(float* __restrict__ v) {
    v[blockIdx.x * 256 + threadIdx.x] = 1.0f;
}

// ---------------------------------------------------------------------------
// K1: S[n][m] = exp(dot(A_n, B_m) / eps)   via WMMA f16 16x16x32
// grid (512 colTiles, 64), block 256 (8 waves); wave -> one 16x16 tile
__global__ void sink_gemm_exp(const float* __restrict__ A,
                              const float* __restrict__ Bm,
                              float* __restrict__ S) {
    const int lane  = threadIdx.x & 31;
    const int wave  = threadIdx.x >> 5;
    const int tileN = blockIdx.x;
    const int tileM = blockIdx.y * 8 + wave;
    const int r  = lane & 15;
    const int hi = lane >> 4;

    const int rowA = tileM * 16 + r;   // A-frag: lane holds row rowA
    const int rowB = tileN * 16 + r;   // B-frag: lane holds column -> row of Bm

    v8f acc = {};
#pragma unroll
    for (int k0 = 0; k0 < DD; k0 += 32) {
        v16h a, b;
        // A-frag K pattern: {hi*8+e, 16+hi*8+e}; groups are 16B-aligned
        const float* ap = A  + (size_t)rowA * DD + k0 + hi * 8;
        const float4 a0 = *(const float4*)(ap);
        const float4 a1 = *(const float4*)(ap + 4);
        const float4 a2 = *(const float4*)(ap + 16);
        const float4 a3 = *(const float4*)(ap + 20);
        a[0] = (_Float16)a0.x;  a[1] = (_Float16)a0.y;  a[2]  = (_Float16)a0.z;  a[3]  = (_Float16)a0.w;
        a[4] = (_Float16)a1.x;  a[5] = (_Float16)a1.y;  a[6]  = (_Float16)a1.z;  a[7]  = (_Float16)a1.w;
        a[8] = (_Float16)a2.x;  a[9] = (_Float16)a2.y;  a[10] = (_Float16)a2.z;  a[11] = (_Float16)a2.w;
        a[12] = (_Float16)a3.x; a[13] = (_Float16)a3.y; a[14] = (_Float16)a3.z;  a[15] = (_Float16)a3.w;
        // B-frag K pattern: {hi*8+e, 16+hi*8+e}
        const float* bp = Bm + (size_t)rowB * DD + k0 + hi * 8;
#pragma unroll
        for (int q = 0; q < 4; ++q) {
            const float4 bq = *(const float4*)(bp + (q & 1) * 4 + (q >> 1) * 16);
            b[4 * q + 0] = (_Float16)bq.x;
            b[4 * q + 1] = (_Float16)bq.y;
            b[4 * q + 2] = (_Float16)bq.z;
            b[4 * q + 3] = (_Float16)bq.w;
        }
        acc = __builtin_amdgcn_wmma_f32_16x16x32_f16(false, a, false, b,
                                                     (short)0, acc, false, false);
    }

    const int n = tileN * 16 + r;
#pragma unroll
    for (int rr = 0; rr < 8; ++rr) {
        const int m = tileM * 16 + rr + 8 * hi;
        S[(size_t)m * NN + n] = __expf(acc[rr] * EPS_INV);
    }
}

// ---------------------------------------------------------------------------
// K1b: pre-pack B fragments (f16) for the final GEMM.
// Layout: Bfrag[chunk][dt][lane][16], contiguous 16 halves per lane (32B).
__global__ void sink_build_bfrag(const float* __restrict__ Bm,
                                 _Float16* __restrict__ Bfrag) {
    const int lane  = threadIdx.x & 31;
    const int dt    = blockIdx.x & 3;
    const int chunk = blockIdx.x >> 2;
    const int hi  = lane >> 4;
    const int col = dt * 16 + (lane & 15);
    const int mb  = chunk * 32 + hi * 8;            // B-frag K pattern: {hi*8+e, 16+hi*8+e}
    _Float16* out = Bfrag + (((size_t)chunk * 4 + dt) * 32 + lane) * 16;
#pragma unroll
    for (int e = 0; e < 16; ++e)
        out[e] = (_Float16)Bm[(size_t)(mb + e + (e >= 8 ? 8 : 0)) * DD + col];
}

// ---------------------------------------------------------------------------
// K2: partialT[slice][m] = sum over slice rows of S[n][m]*v[n].
// The 1KB v-slice is staged into LDS with an async-to-LDS copy (ASYNCcnt).
__global__ void sink_colsum(const float* __restrict__ S,
                            const float* __restrict__ v,
                            float* __restrict__ partialT) {
    __shared__ float vsh[ROWS_PER_SLICE];
    const int m  = blockIdx.x * 256 + threadIdx.x;
    const int sl = blockIdx.y;
    const int n0 = sl * ROWS_PER_SLICE;

    // async copy v[n0 .. n0+255] -> LDS (one b32 per lane), then wait+barrier
    {
        unsigned lds = (unsigned)(size_t)(&vsh[threadIdx.x]);
        const float* gsrc = v + n0 + threadIdx.x;
        asm volatile("global_load_async_to_lds_b32 %0, %1, off"
                     :: "v"(lds), "v"(gsrc) : "memory");
        asm volatile("s_wait_asynccnt 0x0" ::: "memory");
    }
    __syncthreads();

    const float* p = S + (size_t)n0 * NN + m;
    float acc = 0.0f;
#pragma unroll 4
    for (int i = 0; i < ROWS_PER_SLICE; ++i) {
        __builtin_prefetch(p + (size_t)(i + 8) * NN, 0, 0); // global_prefetch_b8
        acc += p[(size_t)i * NN] * vsh[i];
    }
    partialT[(size_t)sl * NN + m] = acc;
}

// K2b: u[m] = 1 / (K * sum_slices partialT[s][m])
__global__ void sink_make_u(const float* __restrict__ partialT,
                            float* __restrict__ u) {
    const int m = blockIdx.x * 256 + threadIdx.x;
    float t = 0.0f;
#pragma unroll
    for (int s = 0; s < SLICES; ++s) t += partialT[(size_t)s * NN + m];
    u[m] = 1.0f / ((float)NN * t);
}

// K3: v[n] = 1 / (B * sum_m S[n][m]*u[m])   (wave per row, float4 lane-strided)
__global__ void sink_rowsum_make_v(const float* __restrict__ S,
                                   const float* __restrict__ u,
                                   float* __restrict__ v) {
    const int lane = threadIdx.x & 31;
    const int wave = threadIdx.x >> 5;
    const int n = blockIdx.x * 8 + wave;
    const float4* row4 = (const float4*)(S + (size_t)n * NN);
    const float4* u4   = (const float4*)u;
    float acc = 0.0f;
#pragma unroll 2
    for (int m = lane; m < NN / 4; m += 32) {
        const float4 sv = row4[m];
        const float4 uv = u4[m];
        acc += sv.x * uv.x + sv.y * uv.y + sv.z * uv.z + sv.w * uv.w;
    }
#pragma unroll
    for (int off = 16; off > 0; off >>= 1) acc += __shfl_xor(acc, off, 32);
    if (lane == 0) v[n] = 1.0f / ((float)NN * acc);
}

// ---------------------------------------------------------------------------
// K4: partialY[seg][n][d] = sum over seg's m-range of (S[n][m]*u[m]*alpha) * Bm[m][d]
// alpha = NN*NN keeps S*u ~ 1.0 in f16 range. One wave per (16-row tile, seg).
__global__ void sink_final_gemm(const float* __restrict__ S,
                                const float* __restrict__ u,
                                const _Float16* __restrict__ Bfrag,
                                float* __restrict__ partialY) {
    const int lane  = threadIdx.x & 31;
    const int tileM = blockIdx.x;
    const int seg   = blockIdx.y;
    const int r  = lane & 15;
    const int hi = lane >> 4;
    const int row = tileM * 16 + r;
    const float alpha = (float)NN * (float)NN;

    const float* srow = S + (size_t)row * NN;
    v8f acc0 = {}, acc1 = {}, acc2 = {}, acc3 = {};

    for (int c = seg * (CHUNKS / SEGS); c < (seg + 1) * (CHUNKS / SEGS); ++c) {
        const int mb = c * 32 + hi * 8;            // A-frag K pattern (32B aligned)
        const float4 s0 = *(const float4*)(srow + mb);
        const float4 s1 = *(const float4*)(srow + mb + 4);
        const float4 s2 = *(const float4*)(srow + mb + 16);
        const float4 s3 = *(const float4*)(srow + mb + 20);
        const float4 u0 = *(const float4*)(u + mb);
        const float4 u1 = *(const float4*)(u + mb + 4);
        const float4 u2 = *(const float4*)(u + mb + 16);
        const float4 u3 = *(const float4*)(u + mb + 20);
        v16h a;
        a[0]  = (_Float16)(s0.x * (u0.x * alpha)); a[1]  = (_Float16)(s0.y * (u0.y * alpha));
        a[2]  = (_Float16)(s0.z * (u0.z * alpha)); a[3]  = (_Float16)(s0.w * (u0.w * alpha));
        a[4]  = (_Float16)(s1.x * (u1.x * alpha)); a[5]  = (_Float16)(s1.y * (u1.y * alpha));
        a[6]  = (_Float16)(s1.z * (u1.z * alpha)); a[7]  = (_Float16)(s1.w * (u1.w * alpha));
        a[8]  = (_Float16)(s2.x * (u2.x * alpha)); a[9]  = (_Float16)(s2.y * (u2.y * alpha));
        a[10] = (_Float16)(s2.z * (u2.z * alpha)); a[11] = (_Float16)(s2.w * (u2.w * alpha));
        a[12] = (_Float16)(s3.x * (u3.x * alpha)); a[13] = (_Float16)(s3.y * (u3.y * alpha));
        a[14] = (_Float16)(s3.z * (u3.z * alpha)); a[15] = (_Float16)(s3.w * (u3.w * alpha));

        const _Float16* bb = Bfrag + ((size_t)c * 4 * 32 + lane) * 16;
        v16h b0 = *(const v16h*)(bb);
        v16h b1 = *(const v16h*)(bb + 32 * 16);
        v16h b2 = *(const v16h*)(bb + 2 * 32 * 16);
        v16h b3 = *(const v16h*)(bb + 3 * 32 * 16);
        acc0 = __builtin_amdgcn_wmma_f32_16x16x32_f16(false, a, false, b0, (short)0, acc0, false, false);
        acc1 = __builtin_amdgcn_wmma_f32_16x16x32_f16(false, a, false, b1, (short)0, acc1, false, false);
        acc2 = __builtin_amdgcn_wmma_f32_16x16x32_f16(false, a, false, b2, (short)0, acc2, false, false);
        acc3 = __builtin_amdgcn_wmma_f32_16x16x32_f16(false, a, false, b3, (short)0, acc3, false, false);
    }

    float* out = partialY + (size_t)seg * NN * DD + (size_t)(tileM * 16) * DD;
#pragma unroll
    for (int rr = 0; rr < 8; ++rr) {
        const int mr = rr + 8 * hi;
        out[(size_t)mr * DD +  0 + r] = acc0[rr];
        out[(size_t)mr * DD + 16 + r] = acc1[rr];
        out[(size_t)mr * DD + 32 + r] = acc2[rr];
        out[(size_t)mr * DD + 48 + r] = acc3[rr];
    }
}

// ---------------------------------------------------------------------------
// K5: per-block squared-error partials (float4; each float4 lies in one row n).
// aligned[n][d] = B * v[n] * Yraw / alpha = v[n] * Yraw / NN
__global__ void sink_loss_partial(const float* __restrict__ partialY,
                                  const float* __restrict__ v,
                                  const float* __restrict__ A,
                                  float* __restrict__ partialLoss) {
    __shared__ float red[256];
    const int t = threadIdx.x;
    const size_t i4 = (size_t)blockIdx.x * 256 + t;   // float4 index, 131072 total
    const float4* pY = (const float4*)partialY;
    const float4* pA = (const float4*)A;

    float4 y = make_float4(0.f, 0.f, 0.f, 0.f);
#pragma unroll
    for (int s = 0; s < SEGS; ++s) {
        const float4 q = pY[(size_t)s * (NN * DD / 4) + i4];
        y.x += q.x; y.y += q.y; y.z += q.z; y.w += q.w;
    }
    const int n = (int)(i4 >> 4);                     // 16 float4 per row of DD=64
    const float sc = v[n] * (1.0f / (float)NN);
    const float4 a = pA[i4];
    const float dx = y.x * sc - a.x;
    const float dy = y.y * sc - a.y;
    const float dz = y.z * sc - a.z;
    const float dw = y.w * sc - a.w;
    red[t] = dx * dx + dy * dy + dz * dz + dw * dw;
    __syncthreads();
    for (int off = 128; off > 0; off >>= 1) {
        if (t < off) red[t] += red[t + off];
        __syncthreads();
    }
    if (t == 0) partialLoss[blockIdx.x] = red[0];
}

// K6: final scalar
__global__ void sink_loss_final(const float* __restrict__ partialLoss,
                                float* __restrict__ out) {
    __shared__ float red[256];
    const int t = threadIdx.x;
    red[t] = partialLoss[t] + partialLoss[t + 256];
    __syncthreads();
    for (int off = 128; off > 0; off >>= 1) {
        if (t < off) red[t] += red[t + off];
        __syncthreads();
    }
    if (t == 0) out[0] = red[0] / ((float)NN * (float)DD);
}

// ---------------------------------------------------------------------------
extern "C" void kernel_launch(void* const* d_in, const int* in_sizes, int n_in,
                              void* d_out, int out_size, void* d_ws, size_t ws_size,
                              hipStream_t stream) {
    const float* A  = (const float*)d_in[0];   // cl_seq2intents [8192,64]
    const float* Bm = (const float*)d_in[1];   // seq2intents    [8192,64]
    float* out = (float*)d_out;

    char* ws = (char*)d_ws;
    size_t off = 0;
    float* S = (float*)(ws + off);            off += (size_t)NN * NN * sizeof(float);       // 256 MB
    float* partialT = (float*)(ws + off);     off += (size_t)SLICES * NN * sizeof(float);   // 1 MB
    float* u = (float*)(ws + off);            off += (size_t)NN * sizeof(float);
    float* v = (float*)(ws + off);            off += (size_t)NN * sizeof(float);
    _Float16* Bfrag = (_Float16*)(ws + off);  off += (size_t)CHUNKS * 4 * 32 * 16 * 2;      // 1 MB
    float* partialY = (float*)(ws + off);     off += (size_t)SEGS * NN * DD * sizeof(float);// 16 MB
    float* partialLoss = (float*)(ws + off);  off += 512 * sizeof(float);

    // init + build S and packed B fragments
    sink_init_v<<<NN / 256, 256, 0, stream>>>(v);
    sink_gemm_exp<<<dim3(NN / 16, NN / 16 / 8), 256, 0, stream>>>(A, Bm, S);
    sink_build_bfrag<<<CHUNKS * 4, 32, 0, stream>>>(Bm, Bfrag);

    // 20 Sinkhorn iterations on the diagonal scalings (u, v)
    for (int it = 0; it < ITERS; ++it) {
        sink_colsum<<<dim3(NN / 256, SLICES), 256, 0, stream>>>(S, v, partialT);
        sink_make_u<<<NN / 256, 256, 0, stream>>>(partialT, u);
        sink_rowsum_make_v<<<NN / 8, 256, 0, stream>>>(S, u, v);
    }

    // aligned = (B * v_n) * (S diag(u)) * Bm, then MSE vs A
    sink_final_gemm<<<dim3(NN / 16, SEGS), 32, 0, stream>>>(S, u, Bfrag, partialY);
    sink_loss_partial<<<NN * DD / 1024, 256, 0, stream>>>(partialY, v, A, partialLoss);
    sink_loss_final<<<1, 256, 0, stream>>>(partialLoss, out);
}
